// MemoryModule_41257455845951
// MI455X (gfx1250) — matrix operand order repaired and checked by reading, not compiled, not run on hardware
//
#include <hip/hip_runtime.h>

typedef _Float16 half_t;
typedef __attribute__((ext_vector_type(16))) _Float16 v16h;
typedef __attribute__((ext_vector_type(8)))  _Float16 v8h;
typedef __attribute__((ext_vector_type(8)))  float    v8f;
typedef __attribute__((ext_vector_type(4)))  unsigned int u32x4;
typedef __attribute__((ext_vector_type(8)))  int i32x8;
typedef __attribute__((ext_vector_type(4)))  int i32x4;

#define SHRINK_T 0.0025f
#define SHRINK_EPS 1e-12f
#define NORM_EPS 1e-12f
#define KTILE 64  // K-chunk staged in LDS per double-buffer slot

__device__ __forceinline__ float wred_sum(float v) {
#pragma unroll
  for (int m = 16; m > 0; m >>= 1) v += __shfl_xor(v, m, 32);
  return v;
}
__device__ __forceinline__ float wred_max(float v) {
#pragma unroll
  for (int m = 16; m > 0; m >>= 1) v = fmaxf(v, __shfl_xor(v, m, 32));
  return v;
}

// ---- Tensor Data Mover: DMA a 16-row x KTILE-half f16 tile (row pitch = strideK
// halves) from global into LDS at byte offset ldsOff. D# per ISA 08_async_tensor §8.
__device__ __forceinline__ void tdm_load_tile_a(const half_t* gptr, unsigned ldsOff,
                                                int strideK) {
  unsigned long long ga = (unsigned long long)(uintptr_t)gptr;
  u32x4 g0;
  g0[0] = 1u;                                    // count=1 (valid user descriptor)
  g0[1] = ldsOff;                                // lds_addr (bytes)
  g0[2] = (unsigned)(ga & 0xFFFFFFFFu);          // global_addr[31:0]
  g0[3] = (unsigned)((ga >> 32) & 0x01FFFFFFu)   // global_addr[56:32]
          | (2u << 30);                          // type = 2 ("image")
  i32x8 g1;
  g1[0] = (1 << 16);                             // wg_mask=0, data_size=1 (2 bytes)
  g1[1] = (int)(((unsigned)strideK & 0xFFFFu) << 16);  // tensor_dim0[15:0]=strideK
  g1[2] = (int)((((unsigned)strideK >> 16) & 0xFFFFu)  // tensor_dim0[31:16]
          | (16u << 16));                        // tensor_dim1[15:0] = 16 rows
  g1[3] = (int)((unsigned)KTILE << 16);          // tensor_dim1[31:16]=0, tile_dim0=KTILE
  g1[4] = 16;                                    // tile_dim1 = 16 rows, tile_dim2 = 0
  g1[5] = strideK;                               // tensor_dim0_stride[31:0]
  g1[6] = 0;                                     // stride[47:32]=0, dim1_stride[15:0]=0
  g1[7] = 0;                                     // dim1_stride[47:16]=0
  i32x4 gz; gz[0] = 0; gz[1] = 0; gz[2] = 0; gz[3] = 0;  // groups 2/3 unused (2D)
#if __clang_major__ >= 23
  i32x8 gz8;
#pragma unroll
  for (int j = 0; j < 8; ++j) gz8[j] = 0;
  __builtin_amdgcn_tensor_load_to_lds(g0, g1, gz, gz, gz8, 0);
#else
  __builtin_amdgcn_tensor_load_to_lds(g0, g1, gz, gz, 0);
#endif
}

// L2-normalize memory rows -> nm16 [M][C]; transpose raw memory -> memT16 [C][M]
__global__ void prep_mem_kernel(const float* __restrict__ mem,
                                half_t* __restrict__ nm16,
                                half_t* __restrict__ memT16,
                                int M, int C) {
  const int lane = threadIdx.x & 31;
  const int wave = threadIdx.x >> 5;
  const int row = blockIdx.x * (blockDim.x >> 5) + wave;
  if (row >= M) return;
  const float* r = mem + (size_t)row * C;
  float ss = 0.f;
  for (int c = lane; c < C; c += 32) { float v = r[c]; ss += v * v; }
  ss = wred_sum(ss);
  const float inv = 1.0f / fmaxf(sqrtf(ss), NORM_EPS);
  for (int c = lane; c < C; c += 32) {
    float v = r[c];
    nm16[(size_t)row * C + c]   = (half_t)(v * inv);
    memT16[(size_t)c * M + row] = (half_t)v;
  }
}

// L2-normalize feature rows -> nf16 [N][C]; copy raw f into concat output cols [C, 2C)
__global__ void prep_feat_kernel(const float* __restrict__ F,
                                 half_t* __restrict__ nf16,
                                 float* __restrict__ outBase,  // row stride 2C
                                 int N, int C) {
  const int lane = threadIdx.x & 31;
  const int wave = threadIdx.x >> 5;
  const int row = blockIdx.x * (blockDim.x >> 5) + wave;
  if (row >= N) return;
  const float* r = F + (size_t)row * C;
  float ss = 0.f;
  for (int c = lane; c < C; c += 32) { float v = r[c]; ss += v * v; }
  ss = wred_sum(ss);
  const float inv = 1.0f / fmaxf(sqrtf(ss), NORM_EPS);
  for (int c = lane; c < C; c += 32) {
    float v = r[c];
    nf16[(size_t)row * C + c] = (half_t)(v * inv);
    outBase[(size_t)row * (2 * C) + C + c] = v;
  }
}

// D[r][c] = sum_k A[r][k] * Bt[c][k]
// A: [rows][K] f16; Bt: [cols][K] f16; D: fp32, row stride ldo.
// Block = 4 waves = 16 rows x 256 cols; wave tile 16x64 (4 accumulators).
// A tile (16 x KTILE) staged in LDS by the TDM, double-buffered; B streamed
// from global with two 16B fragment loads per lane. K multiple of KTILE.
__global__ void gemm_nt_kernel(const half_t* __restrict__ A,
                               const half_t* __restrict__ Bt,
                               float* __restrict__ D,
                               int K, int ldo) {
  __shared__ half_t atile[2][16 * KTILE];
  const int lane   = threadIdx.x & 31;
  const int wave   = threadIdx.x >> 5;
  const int half16 = lane >> 4;
  const int mn     = lane & 15;
  const int row0 = blockIdx.y * 16;
  const int col0 = (blockIdx.x * 4 + wave) * 64;

  v8f acc[4] = {};
  const half_t* brow0 = Bt + (size_t)(col0 + mn) * K;
  const half_t* arow0 = A + (size_t)row0 * K;
  // Flat LDS address truncates to the LDS byte offset (ISA §10.2)
  const unsigned ldsOff0 = (unsigned)(uintptr_t)&atile[0][0];

  const int nChunks = K / KTILE;
  if (wave == 0) tdm_load_tile_a(arow0, ldsOff0, K);  // prologue: chunk 0

  for (int kc = 0; kc < nChunks; ++kc) {
    if (wave == 0) __builtin_amdgcn_s_wait_tensorcnt((short)0);  // chunk kc landed
    __syncthreads();
    if (wave == 0 && kc + 1 < nChunks)  // overlap DMA of kc+1 with compute on kc
      tdm_load_tile_a(arow0 + (size_t)(kc + 1) * KTILE,
                      ldsOff0 + (unsigned)(((kc + 1) & 1) * (16 * KTILE * 2)), K);

    const half_t* at = &atile[kc & 1][0];
    const half_t* bk = brow0 + (size_t)kc * KTILE;
#pragma unroll
    for (int k0 = 0; k0 < KTILE; k0 += 32) {
      if (kc + 1 < nChunks)
        __builtin_prefetch(bk + KTILE + k0, 0, 3);  // next B chunk, WGP scope
      // A fragment from LDS (16-bit 16x32 layout): 2 x ds_load_b128 per lane
      v8h alo = *(const v8h*)(at + mn * KTILE + k0 + 8 * half16);
      v8h ahi = *(const v8h*)(at + mn * KTILE + k0 + 16 + 8 * half16);
      v16h a;
#pragma unroll
      for (int j = 0; j < 8; ++j) { a[j] = alo[j]; a[8 + j] = ahi[j]; }
      // Batch all 4 B fragments, then issue 4 back-to-back WMMAs
      v16h b[4];
#pragma unroll
      for (int t = 0; t < 4; ++t) {
        v8h blo = *(const v8h*)(bk + (size_t)t * 16 * K + k0 + 8 * half16);
        v8h bhi = *(const v8h*)(bk + (size_t)t * 16 * K + k0 + 16 + 8 * half16);
#pragma unroll
        for (int j = 0; j < 8; ++j) { b[t][j] = blo[j]; b[t][8 + j] = bhi[j]; }
      }
#pragma unroll
      for (int t = 0; t < 4; ++t)
        acc[t] = __builtin_amdgcn_wmma_f32_16x16x32_f16(
            false, a, false, b[t], (short)0, acc[t], false, false);
    }
    __syncthreads();  // everyone done reading atile[kc&1] before it is refilled
  }

  // C/D 32-bit 16x16 layout: VGPR r -> M = r + 8*half16, N = mn
#pragma unroll
  for (int t = 0; t < 4; ++t)
#pragma unroll
    for (int r = 0; r < 8; ++r)
      D[(size_t)(row0 + 8 * half16 + r) * ldo + col0 + 16 * t + mn] = acc[t][r];
}

// Row softmax + hard-shrink + L1 normalize over M=1024 cols, in-place fp32,
// plus f16 copy for GEMM2. One wave per row, 32 columns per lane in registers.
__global__ void softmax_shrink_kernel(float* __restrict__ W,
                                      half_t* __restrict__ W16,
                                      int N) {
  const int M = 1024;
  const int lane = threadIdx.x & 31;
  const int wave = threadIdx.x >> 5;
  const int row = blockIdx.x * (blockDim.x >> 5) + wave;
  if (row >= N) return;
  float* wr = W + (size_t)row * M;

  float x[32];
  float mx = -3.402823466e38f;
#pragma unroll
  for (int j = 0; j < 32; ++j) { x[j] = wr[lane + 32 * j]; mx = fmaxf(mx, x[j]); }
  mx = wred_max(mx);

  float s = 0.f;
#pragma unroll
  for (int j = 0; j < 32; ++j) { x[j] = __expf(x[j] - mx); s += x[j]; }
  s = wred_sum(s);
  const float invs = 1.0f / s;

  float l1 = 0.f;
#pragma unroll
  for (int j = 0; j < 32; ++j) {
    float w = x[j] * invs;
    float r = w - SHRINK_T;
    float v = (r > 0.f) ? (r * w) / (r + SHRINK_EPS) : 0.f;  // relu(w-t)*w/(|w-t|+eps)
    x[j] = v;
    l1 += v;  // v >= 0, so L1 sum == plain sum
  }
  l1 = wred_sum(l1);
  const float invl1 = 1.0f / fmaxf(l1, NORM_EPS);

#pragma unroll
  for (int j = 0; j < 32; ++j) {
    float v = x[j] * invl1;
    wr[lane + 32 * j] = v;
    W16[(size_t)row * M + lane + 32 * j] = (half_t)v;
  }
}

extern "C" void kernel_launch(void* const* d_in, const int* in_sizes, int n_in,
                              void* d_out, int out_size, void* d_ws, size_t ws_size,
                              hipStream_t stream) {
  (void)in_sizes; (void)n_in; (void)out_size; (void)ws_size;
  const int N = 16384, M = 1024;
  const int chans[3] = {256, 512, 1024};
  // setup_inputs dict order: feat0, mem0, feat1, mem1, feat2, mem2
  const float* feats[3] = {(const float*)d_in[0], (const float*)d_in[2], (const float*)d_in[4]};
  const float* mems[3]  = {(const float*)d_in[1], (const float*)d_in[3], (const float*)d_in[5]};
  float* out = (float*)d_out;

  // Output layout: out0, out1, out2 (each N x 2C), then w0, w1, w2 (each N x M)
  size_t outOff[3], wOff[3], o = 0;
  for (int i = 0; i < 3; ++i) { outOff[i] = o; o += (size_t)N * 2 * chans[i]; }
  for (int i = 0; i < 3; ++i) { wOff[i] = o; o += (size_t)N * M; }

  // Workspace (levels are stream-serialized, so buffers are reused): ~68 MB
  char* ws = (char*)d_ws;
  half_t* nf16   = (half_t*)ws; ws += (size_t)N * 1024 * sizeof(half_t);
  half_t* w16    = (half_t*)ws; ws += (size_t)N * M * sizeof(half_t);
  half_t* nm16   = (half_t*)ws; ws += (size_t)M * 1024 * sizeof(half_t);
  half_t* memT16 = (half_t*)ws;

  for (int i = 0; i < 3; ++i) {
    const int C = chans[i];
    float* outBase = out + outOff[i];  // [N][2C]
    float* wBase   = out + wOff[i];    // [N][M]

    prep_mem_kernel <<<M / 8, 256, 0, stream>>>(mems[i], nm16, memT16, M, C);
    prep_feat_kernel<<<N / 8, 256, 0, stream>>>(feats[i], nf16, outBase, N, C);

    // GEMM1: sims = nf16 [NxC] @ nm16^T -> fp32 straight into w region of d_out
    gemm_nt_kernel<<<dim3(M / 256, N / 16), 128, 0, stream>>>(nf16, nm16, wBase, C, M);

    softmax_shrink_kernel<<<N / 8, 256, 0, stream>>>(wBase, w16, N);

    // GEMM2: mf = w16 [NxM] @ memT16^T -> concat output cols [0, C)
    gemm_nt_kernel<<<dim3(C / 256, N / 16), 128, 0, stream>>>(w16, memT16, outBase, M, 2 * C);
  }
}
